// SelfAttention_67748814127590
// MI455X (gfx1250) — matrix operand order repaired and checked by reading, not compiled, not run on hardware
//
#include <hip/hip_runtime.h>
#include <math.h>

#define S_LEN 2048
#define D_DIM 1024
#define NBATCH 4
#define NTOK (NBATCH * S_LEN) /* 8192 */

typedef __attribute__((ext_vector_type(16))) __bf16 v16bf;
typedef __attribute__((ext_vector_type(8)))  __bf16 v8bf;
typedef __attribute__((ext_vector_type(8)))  float  v8f;
typedef __attribute__((ext_vector_type(4)))  float  f4;

// ---------------------------------------------------------------------------
// CDNA5 WMMA 16x16x32 bf16 fragment layouts (CDNA5 ISA):
//  A (16x32): lane holds row M=lane%16; lanes 0-15: K=0..7 & 16..23,
//             lanes 16-31: K=8..15 & 24..31  -> two contiguous 8-elem runs.
//  B (32x16): lane holds col N=lane%16; lanes 0-15: K=0..7 & 16..23,
//             lanes 16-31: K=8..15 & 24..31  -> two contiguous 8-elem runs.
//  C/D (16x16 f32): VGPR r: lanes 0-15 M=r, lanes 16-31 M=8+r; N=lane%16.
// ---------------------------------------------------------------------------

__device__ __forceinline__ v16bf cat8(v8bf lo, v8bf hi) {
  return __builtin_shufflevector(lo, hi, 0, 1, 2, 3, 4, 5, 6, 7,
                                         8, 9, 10, 11, 12, 13, 14, 15);
}

__device__ __forceinline__ v16bf load_a_bf(const __bf16* __restrict__ base,
                                           int ld, int row, int kb, int half) {
  const __bf16* p = base + (size_t)row * ld + kb + half * 8;
  v8bf lo = *(const v8bf*)p;          // K run 1 (16B aligned)
  v8bf hi = *(const v8bf*)(p + 16);   // K run 2
  return cat8(lo, hi);
}

__device__ __forceinline__ v16bf load_b_bf(const __bf16* __restrict__ base,
                                           int ld, int srcrow, int kb, int half) {
  const __bf16* p = base + (size_t)srcrow * ld + kb + half * 8;
  v8bf lo = *(const v8bf*)p;
  v8bf hi = *(const v8bf*)(p + 16);
  return cat8(lo, hi);
}

#define WMMA_BF16(A, B, C) \
  __builtin_amdgcn_wmma_f32_16x16x32_bf16(false, (A), false, (B), (short)0, (C), false, false)

// ---------------------------------------------------------------------------
// Kernel 0: one-shot fp32 -> bf16 conversion (vectorized, grid-stride).
// ---------------------------------------------------------------------------
__global__ __launch_bounds__(256) void cvt_kernel(
    const float* __restrict__ src, __bf16* __restrict__ dst, int n4) {
  const int stride = gridDim.x * blockDim.x;
  for (int i = blockIdx.x * blockDim.x + threadIdx.x; i < n4; i += stride) {
    f4 v = ((const f4*)src)[i];
    __bf16 o0 = (__bf16)v.x, o1 = (__bf16)v.y, o2 = (__bf16)v.z, o3 = (__bf16)v.w;
    typedef __attribute__((ext_vector_type(4))) __bf16 v4bf;
    v4bf o = {o0, o1, o2, o3};
    ((v4bf*)dst)[i] = o;
  }
}

// ---------------------------------------------------------------------------
// Kernel 1: Q/K/V = x @ W^T, all-bf16 fragments.  Per wave: 32M x 64N
// (2 A-frags x 4 B-frags, 8 WMMAs per K-step).  Block = 8 waves, 128x128.
// Q,K stored row-major bf16; V stored transposed Vt[b][d][s].
// Destination-select branch hoisted out of the store loops (block-uniform).
// ---------------------------------------------------------------------------
__global__ __launch_bounds__(256) void qkv_kernel(
    const __bf16* __restrict__ xb, const __bf16* __restrict__ Wb,
    __bf16* __restrict__ Qbf, __bf16* __restrict__ Kbf,
    __bf16* __restrict__ Vt) {
  const int lane = threadIdx.x & 31;
  const int wave = threadIdx.x >> 5;
  const int half = lane >> 4;
  const int lrow = lane & 15;
  const int wm = wave & 3;
  const int wn = wave >> 2;
  const int mq0 = blockIdx.x * 128 + wm * 32;
  const int nb  = blockIdx.y * 128 + wn * 64;
  const int which = blockIdx.z;
  const __bf16* W = Wb + (size_t)which * D_DIM * D_DIM;  // Wq|Wk|Wv packed

  v8f acc[2][4] = {};
  for (int kb = 0; kb < D_DIM; kb += 32) {
    v16bf a0 = load_a_bf(xb, D_DIM, mq0 + lrow, kb, half);
    v16bf a1 = load_a_bf(xb, D_DIM, mq0 + 16 + lrow, kb, half);
    v16bf bf[4];
#pragma unroll
    for (int t = 0; t < 4; ++t)
      bf[t] = load_b_bf(W, D_DIM, nb + t * 16 + lrow, kb, half);
#pragma unroll
    for (int t = 0; t < 4; ++t) {
      acc[0][t] = WMMA_BF16(a0, bf[t], acc[0][t]);
      acc[1][t] = WMMA_BF16(a1, bf[t], acc[1][t]);
    }
  }

  // ---- epilogue: branch on `which` ONCE, then tight store loops ----
  const int mrow0 = mq0 + half * 8;  // this lane's first output row
  if (which < 2) {
    __bf16* __restrict__ dst = (which == 0) ? Qbf : Kbf;
    __bf16* p = dst + (size_t)mrow0 * D_DIM + nb + lrow;
#pragma unroll
    for (int i = 0; i < 2; ++i) {
#pragma unroll
      for (int r = 0; r < 8; ++r) {
#pragma unroll
        for (int t = 0; t < 4; ++t)
          p[(size_t)(i * 16 + r) * D_DIM + t * 16] = (__bf16)acc[i][t][r];
      }
    }
  } else {
    // V transposed: Vt[b][d][s], d = nb + t*16 + lrow, s = (m % S)
    const int bidx = mrow0 >> 11;
    const int s0 = mrow0 & (S_LEN - 1);
    __bf16* p = Vt + (size_t)bidx * D_DIM * S_LEN + (size_t)(nb + lrow) * S_LEN + s0;
#pragma unroll
    for (int i = 0; i < 2; ++i) {
#pragma unroll
      for (int r = 0; r < 8; ++r) {
#pragma unroll
        for (int t = 0; t < 4; ++t)
          p[(size_t)(t * 16) * S_LEN + i * 16 + r] = (__bf16)acc[i][t][r];
      }
    }
  }
}

// ---------------------------------------------------------------------------
// Kernel 2: scores = (Q @ K^T)/sqrt(D), causal mask fused.
// Block-level skip above diagonal; wave-level compute skip for masked tiles.
// ---------------------------------------------------------------------------
__global__ __launch_bounds__(256) void scores_kernel(
    const __bf16* __restrict__ Qbf, const __bf16* __restrict__ Kbf,
    float* __restrict__ scores) {
  const int q0 = blockIdx.x * 128;
  const int k0 = blockIdx.y * 128;
  if (k0 > q0 + 127) return;  // fully masked block (uniform)

  const int lane = threadIdx.x & 31;
  const int wave = threadIdx.x >> 5;
  const int half = lane >> 4;
  const int lrow = lane & 15;
  const int wm = wave & 3;
  const int wn = wave >> 2;
  const int b = blockIdx.z;
  const int mq0 = q0 + wm * 32;
  const int nb  = k0 + wn * 64;

  const __bf16* Qb = Qbf + (size_t)b * S_LEN * D_DIM;
  const __bf16* Kb = Kbf + (size_t)b * S_LEN * D_DIM;

  v8f acc[2][4] = {};
  if (nb <= mq0 + 31) {  // wave tile touches the causal region (wave-uniform)
    for (int kb = 0; kb < D_DIM; kb += 32) {
      v16bf a0 = load_a_bf(Qb, D_DIM, mq0 + lrow, kb, half);
      v16bf a1 = load_a_bf(Qb, D_DIM, mq0 + 16 + lrow, kb, half);
      v16bf bf[4];
#pragma unroll
      for (int t = 0; t < 4; ++t)
        bf[t] = load_b_bf(Kb, D_DIM, nb + t * 16 + lrow, kb, half);
#pragma unroll
      for (int t = 0; t < 4; ++t) {
        acc[0][t] = WMMA_BF16(a0, bf[t], acc[0][t]);
        acc[1][t] = WMMA_BF16(a1, bf[t], acc[1][t]);
      }
    }
  }

  float* sb = scores + ((size_t)b * S_LEN + mq0 + half * 8) * S_LEN + nb + lrow;
#pragma unroll
  for (int i = 0; i < 2; ++i) {
#pragma unroll
    for (int r = 0; r < 8; ++r) {
      const int qi = mq0 + i * 16 + half * 8 + r;
#pragma unroll
      for (int t = 0; t < 4; ++t) {
        const int kj = nb + t * 16 + lrow;
        float v = acc[i][t][r] * 0.03125f;  // 1/sqrt(1024)
        if (kj > qi) v = -__builtin_inff();
        sb[(size_t)(i * 16 + r) * S_LEN + t * 16] = v;
      }
    }
  }
}

// ---------------------------------------------------------------------------
// Kernel 3: row-wise causal softmax, fp32 -> bf16 P (zeros above diagonal).
// One wave32 per row; shuffle reductions.
// ---------------------------------------------------------------------------
__global__ __launch_bounds__(256) void softmax_kernel(
    const float* __restrict__ scores, __bf16* __restrict__ P) {
  const int lane = threadIdx.x & 31;
  const int wave = threadIdx.x >> 5;
  const int row = blockIdx.x * 8 + wave;  // 0..8191
  const int b = row >> 11;
  const int i = row & (S_LEN - 1);
  const float* s = scores + ((size_t)b * S_LEN + i) * S_LEN;
  __bf16* p = P + ((size_t)b * S_LEN + i) * S_LEN;
  const int n = i + 1;

  float m = -__builtin_inff();
  for (int j = lane; j < n; j += 32) m = fmaxf(m, s[j]);
#pragma unroll
  for (int off = 16; off; off >>= 1) m = fmaxf(m, __shfl_xor(m, off, 32));

  float sum = 0.0f;
  for (int j = lane; j < n; j += 32) sum += __expf(s[j] - m);
#pragma unroll
  for (int off = 16; off; off >>= 1) sum += __shfl_xor(sum, off, 32);

  const float inv = 1.0f / sum;
  for (int j = lane; j < n; j += 32) p[j] = (__bf16)(__expf(s[j] - m) * inv);
  for (int j = n + lane; j < S_LEN; j += 32) p[j] = (__bf16)0.0f;
}

// ---------------------------------------------------------------------------
// Kernel 4: out = P @ V.  Vt[b][d][s] makes B-fragments contiguous.
// Per-wave K-loop truncated at the wave's diagonal (P zero beyond it).
// ---------------------------------------------------------------------------
__global__ __launch_bounds__(256) void pv_kernel(
    const __bf16* __restrict__ P, const __bf16* __restrict__ Vt,
    float* __restrict__ out) {
  const int lane = threadIdx.x & 31;
  const int wave = threadIdx.x >> 5;
  const int half = lane >> 4;
  const int lrow = lane & 15;
  const int wm = wave & 3;
  const int wn = wave >> 2;
  const int b = blockIdx.z;
  const int q0 = blockIdx.x * 128;
  const int d0 = blockIdx.y * 128;
  const int mq0 = q0 + wm * 32;
  const int nd  = d0 + wn * 64;

  const __bf16* Pb = P + (size_t)b * S_LEN * S_LEN;
  const __bf16* Vb = Vt + (size_t)b * D_DIM * S_LEN;

  const int kend = mq0 + 32;  // causal truncation, wave-uniform
  v8f acc[2][4] = {};
  for (int kt = 0; kt < kend; kt += 32) {
    v16bf a0 = load_a_bf(Pb, S_LEN, mq0 + lrow, kt, half);
    v16bf a1 = load_a_bf(Pb, S_LEN, mq0 + 16 + lrow, kt, half);
    v16bf bf[4];
#pragma unroll
    for (int t = 0; t < 4; ++t)
      bf[t] = load_b_bf(Vb, S_LEN, nd + t * 16 + lrow, kt, half);
#pragma unroll
    for (int t = 0; t < 4; ++t) {
      acc[0][t] = WMMA_BF16(a0, bf[t], acc[0][t]);
      acc[1][t] = WMMA_BF16(a1, bf[t], acc[1][t]);
    }
  }

  float* po = out + ((size_t)b * S_LEN + mq0 + half * 8) * D_DIM + nd + lrow;
#pragma unroll
  for (int i = 0; i < 2; ++i) {
#pragma unroll
    for (int r = 0; r < 8; ++r) {
#pragma unroll
      for (int t = 0; t < 4; ++t)
        po[(size_t)(i * 16 + r) * D_DIM + t * 16] = acc[i][t][r];
    }
  }
}

// ---------------------------------------------------------------------------
extern "C" void kernel_launch(void* const* d_in, const int* in_sizes, int n_in,
                              void* d_out, int out_size, void* d_ws, size_t ws_size,
                              hipStream_t stream) {
  const float* x  = (const float*)d_in[0];
  const float* Wq = (const float*)d_in[1];
  const float* Wk = (const float*)d_in[2];
  const float* Wv = (const float*)d_in[3];
  float* out = (float*)d_out;

  // Workspace layout (bytes):
  //   Qbf   [0,        16 MB)   bf16 [8192,1024]
  //   Kbf   [16 MB,    32 MB)   bf16 [8192,1024]
  //   Vt    [32 MB,    48 MB)   bf16 [4][1024][2048]
  //   score [48 MB,   112 MB)   f32  [4][2048][2048]   (L2-resident)
  //   Pbf   [112 MB,  144 MB)   bf16 [4][2048][2048]
  //   xb    [144 MB,  160 MB)   bf16 [8192,1024]
  //   Wb    [160 MB,  166 MB)   bf16 [3][1024,1024]  (Wq|Wk|Wv)
  char* ws = (char*)d_ws;
  const size_t MB = 1024ull * 1024ull;
  __bf16* Qbf    = (__bf16*)(ws + 0);
  __bf16* Kbf    = (__bf16*)(ws + 16 * MB);
  __bf16* Vt     = (__bf16*)(ws + 32 * MB);
  float*  scores = (float*) (ws + 48 * MB);
  __bf16* Pbf    = (__bf16*)(ws + 112 * MB);
  __bf16* xb     = (__bf16*)(ws + 144 * MB);
  __bf16* Wb     = (__bf16*)(ws + 160 * MB);

  dim3 blk(256);
  // 0) one-shot fp32 -> bf16 conversions
  cvt_kernel<<<dim3(512), blk, 0, stream>>>(x, xb, NTOK * D_DIM / 4);
  cvt_kernel<<<dim3(128), blk, 0, stream>>>(Wq, Wb + 0ull * D_DIM * D_DIM,
                                            D_DIM * D_DIM / 4);
  cvt_kernel<<<dim3(128), blk, 0, stream>>>(Wk, Wb + 1ull * D_DIM * D_DIM,
                                            D_DIM * D_DIM / 4);
  cvt_kernel<<<dim3(128), blk, 0, stream>>>(Wv, Wb + 2ull * D_DIM * D_DIM,
                                            D_DIM * D_DIM / 4);
  // 1) QKV projections
  qkv_kernel<<<dim3(NTOK / 128, D_DIM / 128, 3), blk, 0, stream>>>(
      xb, Wb, Qbf, Kbf, Vt);
  // 2) scaled + masked scores
  scores_kernel<<<dim3(S_LEN / 128, S_LEN / 128, NBATCH), blk, 0, stream>>>(
      Qbf, Kbf, scores);
  // 3) causal softmax
  softmax_kernel<<<dim3(NTOK / 8), blk, 0, stream>>>(scores, Pbf);
  // 4) P @ V
  pv_kernel<<<dim3(S_LEN / 128, D_DIM / 128, NBATCH), blk, 0, stream>>>(
      Pbf, Vt, out);
}
